// ModeAdaptiveLinear_59176059404600
// MI455X (gfx1250) — matrix operand order repaired and checked by reading, not compiled, not run on hardware
//
#include <hip/hip_runtime.h>
#include <stdint.h>

typedef __attribute__((ext_vector_type(16))) __bf16 v16bf;
typedef __attribute__((ext_vector_type(8)))  __bf16 v8bf;
typedef __attribute__((ext_vector_type(4)))  __bf16 v4bf;
typedef __attribute__((ext_vector_type(8)))  float  v8f;

#define N_CTRL 8
#define D_IN   512
#define D_OUT  512
#define BM 128
#define BN 128
#define BK 64
#define LDK 72   // BK + 8 bf16 pad: 144B row stride -> 16B aligned, conflict-free b128

static __device__ __forceinline__ v16bf cat8(v8bf a, v8bf b) {
    return __builtin_shufflevector(a, b, 0,1,2,3,4,5,6,7,8,9,10,11,12,13,14,15);
}

// ---- CDNA5 async global->LDS copy (per-lane 16B), tracked by ASYNCcnt ----
static __device__ __forceinline__ void async_copy16(uint32_t lds_off, const void* gptr) {
    asm volatile("global_load_async_to_lds_b128 %0, %1, off"
                 :: "v"(lds_off), "v"(gptr)
                 : "memory");
}

// =====================  shared building blocks  =====================

// (used by the no-workspace fallback kernel)
static __device__ __forceinline__ void stage_A(const float* __restrict__ X,
                                               const float* __restrict__ G,
                                               __bf16* sAh, __bf16* sAl,
                                               int tileM0, int c, int k0, int tid)
{
    const int rsub = tid >> 4;          // 0..15
    const int kq   = (tid & 15) * 4;    // 0..60
    #pragma unroll
    for (int it = 0; it < 8; ++it) {
        const int r    = it * 16 + rsub;
        const int gRow = tileM0 + r;
        const float4 xv = *(const float4*)(X + (size_t)gRow * D_IN + k0 + kq);
        const float  s  = G[(size_t)gRow * N_CTRL + c];
        float f0 = xv.x * s, f1 = xv.y * s, f2 = xv.z * s, f3 = xv.w * s;
        __bf16 h0 = (__bf16)f0, h1 = (__bf16)f1, h2 = (__bf16)f2, h3 = (__bf16)f3;
        __bf16 l0 = (__bf16)(f0 - (float)h0);
        __bf16 l1 = (__bf16)(f1 - (float)h1);
        __bf16 l2 = (__bf16)(f2 - (float)h2);
        __bf16 l3 = (__bf16)(f3 - (float)h3);
        *(v4bf*)(&sAh[r * LDK + kq]) = (v4bf){h0, h1, h2, h3};
        *(v4bf*)(&sAl[r * LDK + kq]) = (v4bf){l0, l1, l2, l3};
    }
}

// ---- pipelined A path: global->regs, regs->bf16 split regs, regs->LDS ----
static __device__ __forceinline__ void prefetch_X(const float* __restrict__ X,
                                                  const float* __restrict__ G,
                                                  int tileM0, int c, int k0, int tid,
                                                  float4 (&xp)[8], float (&gp)[8])
{
    const int rsub = tid >> 4;
    const int kq   = (tid & 15) * 4;
    #pragma unroll
    for (int it = 0; it < 8; ++it) {
        const int gRow = tileM0 + it * 16 + rsub;
        xp[it] = *(const float4*)(X + (size_t)gRow * D_IN + k0 + kq);
        gp[it] = G[(size_t)gRow * N_CTRL + c];
    }
}

static __device__ __forceinline__ void convert_A(const float4 (&xp)[8], const float (&gp)[8],
                                                 v4bf (&ahp)[8], v4bf (&alp)[8])
{
    #pragma unroll
    for (int it = 0; it < 8; ++it) {
        const float s = gp[it];
        float f0 = xp[it].x * s, f1 = xp[it].y * s, f2 = xp[it].z * s, f3 = xp[it].w * s;
        __bf16 h0 = (__bf16)f0, h1 = (__bf16)f1, h2 = (__bf16)f2, h3 = (__bf16)f3;
        __bf16 l0 = (__bf16)(f0 - (float)h0);
        __bf16 l1 = (__bf16)(f1 - (float)h1);
        __bf16 l2 = (__bf16)(f2 - (float)h2);
        __bf16 l3 = (__bf16)(f3 - (float)h3);
        ahp[it] = (v4bf){h0, h1, h2, h3};
        alp[it] = (v4bf){l0, l1, l2, l3};
    }
}

static __device__ __forceinline__ void store_A(__bf16* sAh, __bf16* sAl,
                                               const v4bf (&ahp)[8], const v4bf (&alp)[8],
                                               int tid)
{
    const int rsub = tid >> 4;
    const int kq   = (tid & 15) * 4;
    #pragma unroll
    for (int it = 0; it < 8; ++it) {
        const int r = it * 16 + rsub;
        *(v4bf*)(&sAh[r * LDK + kq]) = ahp[it];
        *(v4bf*)(&sAl[r * LDK + kq]) = alp[it];
    }
}

static __device__ __forceinline__ void compute_chunk(const __bf16* sAh, const __bf16* sAl,
                                                     const __bf16* sBh, const __bf16* sBl,
                                                     v8f (&acc)[4][2],
                                                     int waveM, int waveN, int lm, int lh)
{
    #pragma unroll
    for (int ks = 0; ks < BK; ks += 32) {
        v16bf ah[4], al[4], bh[2], bl[2];
        #pragma unroll
        for (int mt = 0; mt < 4; ++mt) {
            // A frag (16x32 bf16): lane half lh -> K 0..15 / 16..31
            const int r  = waveM + mt * 16 + lm;
            const int kb = ks + lh * 16;
            v8bf hl  = *(const v8bf*)(&sAh[r * LDK + kb]);
            v8bf hh  = *(const v8bf*)(&sAh[r * LDK + kb + 8]);
            ah[mt] = cat8(hl, hh);
            v8bf ll  = *(const v8bf*)(&sAl[r * LDK + kb]);
            v8bf lhv = *(const v8bf*)(&sAl[r * LDK + kb + 8]);
            al[mt] = cat8(ll, lhv);
        }
        #pragma unroll
        for (int nt = 0; nt < 2; ++nt) {
            // B frag (32x16 bf16): lane half lh -> K 0..15 / 16..31, col = lm
            const int o  = waveN + nt * 16 + lm;
            const int kb = ks + lh * 16;
            v8bf h0 = *(const v8bf*)(&sBh[o * LDK + kb]);
            v8bf h1 = *(const v8bf*)(&sBh[o * LDK + kb + 8]);
            bh[nt] = cat8(h0, h1);
            v8bf l0 = *(const v8bf*)(&sBl[o * LDK + kb]);
            v8bf l1 = *(const v8bf*)(&sBl[o * LDK + kb + 8]);
            bl[nt] = cat8(l0, l1);
        }
        #pragma unroll
        for (int mt = 0; mt < 4; ++mt)
            #pragma unroll
            for (int nt = 0; nt < 2; ++nt) {
                acc[mt][nt] = __builtin_amdgcn_wmma_f32_16x16x32_bf16(
                    false, ah[mt], false, bh[nt], (short)0, acc[mt][nt], false, false);
                acc[mt][nt] = __builtin_amdgcn_wmma_f32_16x16x32_bf16(
                    false, ah[mt], false, bl[nt], (short)0, acc[mt][nt], false, false);
                acc[mt][nt] = __builtin_amdgcn_wmma_f32_16x16x32_bf16(
                    false, al[mt], false, bh[nt], (short)0, acc[mt][nt], false, false);
            }
    }
}

static __device__ __forceinline__ void epilogue(const float* __restrict__ G,
                                                const float* __restrict__ Bv,
                                                float* __restrict__ Out,
                                                float* sG, float* sBias,
                                                v8f (&acc)[4][2],
                                                int tileM0, int tileN0,
                                                int waveM, int waveN, int lm, int lh, int tid)
{
    __syncthreads();
    {
        const int r  = tid >> 1;
        const int c4 = (tid & 1) * 4;
        *(float4*)(&sG[r * N_CTRL + c4]) =
            *(const float4*)(G + (size_t)(tileM0 + r) * N_CTRL + c4);
        const int bc = tid >> 5;
        const int oq = (tid & 31) * 4;
        *(float4*)(&sBias[bc * BN + oq]) =
            *(const float4*)(Bv + (size_t)bc * D_OUT + tileN0 + oq);
    }
    __syncthreads();

    #pragma unroll
    for (int mt = 0; mt < 4; ++mt) {
        #pragma unroll
        for (int nt = 0; nt < 2; ++nt) {
            const int colL = waveN + nt * 16 + lm;
            const int col  = tileN0 + colL;
            #pragma unroll
            for (int g = 0; g < 8; ++g) {
                const int rL = waveM + mt * 16 + lh * 8 + g;  // C frag: VGPR g -> row g (+8 hi half)
                float v = acc[mt][nt][g];
                #pragma unroll
                for (int cc = 0; cc < N_CTRL; ++cc)
                    v += sG[rL * N_CTRL + cc] * sBias[cc * BN + colL];
                Out[(size_t)(tileM0 + rL) * D_OUT + col] = v;
            }
        }
    }
}

// =====================  prep: w -> bf16 hi/lo, transposed [c][o][k]  =====================

__global__ __launch_bounds__(256)
void prep_w_kernel(const float* __restrict__ W, __bf16* __restrict__ WTh,
                   __bf16* __restrict__ WTl)
{
    __shared__ float tile[64][65];
    const int tid = threadIdx.x;
    const int c   = blockIdx.z;
    const int i0  = blockIdx.y * 64;
    const int o0  = blockIdx.x * 64;
    const int r   = tid >> 4;           // 0..15
    const int q   = (tid & 15) * 4;     // 0..60
    const float* src = W + ((size_t)c * D_IN + i0) * D_OUT + o0;
    #pragma unroll
    for (int it = 0; it < 4; ++it) {
        const int i = it * 16 + r;
        const float4 v = *(const float4*)(src + (size_t)i * D_OUT + q);
        tile[i][q + 0] = v.x; tile[i][q + 1] = v.y;
        tile[i][q + 2] = v.z; tile[i][q + 3] = v.w;
    }
    __syncthreads();
    #pragma unroll
    for (int it = 0; it < 4; ++it) {
        const int o = it * 16 + r;
        const float f0 = tile[q + 0][o], f1 = tile[q + 1][o];
        const float f2 = tile[q + 2][o], f3 = tile[q + 3][o];
        __bf16 h0 = (__bf16)f0, h1 = (__bf16)f1, h2 = (__bf16)f2, h3 = (__bf16)f3;
        __bf16 l0 = (__bf16)(f0 - (float)h0);
        __bf16 l1 = (__bf16)(f1 - (float)h1);
        __bf16 l2 = (__bf16)(f2 - (float)h2);
        __bf16 l3 = (__bf16)(f3 - (float)h3);
        const size_t dst = (size_t)(c * D_OUT + o0 + o) * D_IN + i0 + q;
        *(v4bf*)(WTh + dst) = (v4bf){h0, h1, h2, h3};
        *(v4bf*)(WTl + dst) = (v4bf){l0, l1, l2, l3};
    }
}

// =====================  main GEMM, async double-buffered B + pipelined A  =====================

static __device__ __forceinline__ void issue_B_async(const __bf16* __restrict__ WTh,
                                                     const __bf16* __restrict__ WTl,
                                                     __bf16* dsth, __bf16* dstl,
                                                     int tileN0, int c, int k0, int tid)
{
    // tile = 128 o-rows x 64 k (bf16) = 1024 16B chunks per h/l; 4 + 4 per thread
    #pragma unroll
    for (int j = 0; j < 4; ++j) {
        const int id = j * 256 + tid;
        const int o  = id >> 3;
        const int kc = id & 7;
        const size_t grow = (size_t)(c * D_OUT + tileN0 + o) * D_IN + k0 + kc * 8;
        const int    loff = o * LDK + kc * 8;
        async_copy16((uint32_t)(uintptr_t)(dsth + loff), WTh + grow);
        async_copy16((uint32_t)(uintptr_t)(dstl + loff), WTl + grow);
    }
}

__global__ __launch_bounds__(256)
void moe_wmma_async(const float* __restrict__ X, const float* __restrict__ G,
                    const __bf16* __restrict__ WTh, const __bf16* __restrict__ WTl,
                    const float* __restrict__ Bv, float* __restrict__ Out)
{
    __shared__ __bf16 sAh[BM * LDK];
    __shared__ __bf16 sAl[BM * LDK];
    __shared__ __bf16 sB[2][2][BN * LDK];   // [buffer][hi/lo], transposed [o][k]
    __shared__ float  sG[BM * N_CTRL];
    __shared__ float  sBias[N_CTRL * BN];

    const int tid   = threadIdx.x;
    const int lane  = tid & 31;
    const int wv    = tid >> 5;
    const int lm    = lane & 15;
    const int lh    = lane >> 4;
    const int waveM = (wv & 1) * 64;
    const int waveN = (wv >> 1) * 32;
    const int tileN0 = blockIdx.x * BN;
    const int tileM0 = blockIdx.y * BM;

    v8f acc[4][2];
    #pragma unroll
    for (int i = 0; i < 4; ++i)
        #pragma unroll
        for (int j = 0; j < 2; ++j)
            acc[i][j] = (v8f){0.f,0.f,0.f,0.f,0.f,0.f,0.f,0.f};

    float4 xp[8]; float gp[8];
    v4bf   ahp[8], alp[8];

    // prologue: start B(0) DMA immediately, then prepare A(0) in registers
    issue_B_async(WTh, WTl, sB[0][0], sB[0][1], tileN0, 0, 0, tid);
    prefetch_X(X, G, tileM0, 0, 0, tid, xp, gp);
    convert_A(xp, gp, ahp, alp);

    for (int iter = 0; iter < N_CTRL * (D_IN / BK); ++iter) {   // 64
        __syncthreads();                       // prior chunk fully consumed everywhere
        store_A(sAh, sAl, ahp, alp, tid);      // only 16 ds_stores in the serial window
        if (iter + 1 < 64) {
            const int nb = (iter + 1) & 1;
            issue_B_async(WTh, WTl, sB[nb][0], sB[nb][1],
                          tileN0, (iter + 1) >> 3, ((iter + 1) & 7) * BK, tid);
            // async loads complete in order: <=8 outstanding => current buffer resident
            asm volatile("s_wait_asynccnt 0x8" ::: "memory");
        } else {
            asm volatile("s_wait_asynccnt 0x0" ::: "memory");
        }
        __syncthreads();
        if (iter + 1 < 64) {                   // global-load latency hides under WMMAs
            prefetch_X(X, G, tileM0, (iter + 1) >> 3, ((iter + 1) & 7) * BK, tid, xp, gp);
        }
        compute_chunk(sAh, sAl, sB[iter & 1][0], sB[iter & 1][1],
                      acc, waveM, waveN, lm, lh);
        if (iter + 1 < 64) {                   // VALU split co-executes with XDL stream
            convert_A(xp, gp, ahp, alp);
        }
    }

    epilogue(G, Bv, Out, sG, sBias, acc, tileM0, tileN0, waveM, waveN, lm, lh, tid);
}

// =====================  fallback: direct (no workspace)  =====================

__global__ __launch_bounds__(256)
void moe_wmma_direct(const float* __restrict__ X, const float* __restrict__ G,
                     const float* __restrict__ W, const float* __restrict__ Bv,
                     float* __restrict__ Out)
{
    __shared__ __bf16 sAh[BM * LDK];
    __shared__ __bf16 sAl[BM * LDK];
    __shared__ __bf16 sBh[BN * LDK];
    __shared__ __bf16 sBl[BN * LDK];
    __shared__ float  sG[BM * N_CTRL];
    __shared__ float  sBias[N_CTRL * BN];

    const int tid   = threadIdx.x;
    const int lane  = tid & 31;
    const int wv    = tid >> 5;
    const int lm    = lane & 15;
    const int lh    = lane >> 4;
    const int waveM = (wv & 1) * 64;
    const int waveN = (wv >> 1) * 32;
    const int tileN0 = blockIdx.x * BN;
    const int tileM0 = blockIdx.y * BM;

    v8f acc[4][2];
    #pragma unroll
    for (int i = 0; i < 4; ++i)
        #pragma unroll
        for (int j = 0; j < 2; ++j)
            acc[i][j] = (v8f){0.f,0.f,0.f,0.f,0.f,0.f,0.f,0.f};

    for (int c = 0; c < N_CTRL; ++c) {
        for (int k0 = 0; k0 < D_IN; k0 += BK) {
            __syncthreads();
            stage_A(X, G, sAh, sAl, tileM0, c, k0, tid);
            {
                const int ksub = tid >> 5;
                const int oq   = (tid & 31) * 4;
                const float* wbase = W + ((size_t)c * D_IN + k0) * D_OUT + tileN0;
                #pragma unroll
                for (int it = 0; it < 8; ++it) {
                    const int kr = it * 8 + ksub;
                    const float4 wv4 = *(const float4*)(wbase + (size_t)kr * D_OUT + oq);
                    const float f[4] = {wv4.x, wv4.y, wv4.z, wv4.w};
                    #pragma unroll
                    for (int j = 0; j < 4; ++j) {
                        __bf16 h = (__bf16)f[j];
                        __bf16 l = (__bf16)(f[j] - (float)h);
                        sBh[(oq + j) * LDK + kr] = h;
                        sBl[(oq + j) * LDK + kr] = l;
                    }
                }
            }
            __syncthreads();
            compute_chunk(sAh, sAl, sBh, sBl, acc, waveM, waveN, lm, lh);
        }
    }
    epilogue(G, Bv, Out, sG, sBias, acc, tileM0, tileN0, waveM, waveN, lm, lh, tid);
}

// =====================  launch  =====================

extern "C" void kernel_launch(void* const* d_in, const int* in_sizes, int n_in,
                              void* d_out, int out_size, void* d_ws, size_t ws_size,
                              hipStream_t stream) {
    const float* X  = (const float*)d_in[0];   // inputs  [B, 512]
    const float* G  = (const float*)d_in[1];   // weights [B, 8]
    const float* W  = (const float*)d_in[2];   // w       [8, 512, 512]
    const float* Bv = (const float*)d_in[3];   // b       [8, 512]
    float* Out = (float*)d_out;

    const int B = in_sizes[0] / D_IN;          // 8192
    dim3 grid(D_OUT / BN, B / BM);             // (4, 64)
    dim3 block(256);

    const size_t WS_NEED = (size_t)2 * N_CTRL * D_IN * D_OUT * sizeof(__bf16); // 8 MB
    if (ws_size >= WS_NEED) {
        __bf16* WTh = (__bf16*)d_ws;
        __bf16* WTl = WTh + (size_t)N_CTRL * D_IN * D_OUT;
        hipLaunchKernelGGL(prep_w_kernel, dim3(D_OUT / 64, D_IN / 64, N_CTRL), block, 0,
                           stream, W, WTh, WTl);
        hipLaunchKernelGGL(moe_wmma_async, grid, block, 0, stream, X, G, WTh, WTl, Bv, Out);
    } else {
        hipLaunchKernelGGL(moe_wmma_direct, grid, block, 0, stream, X, G, W, Bv, Out);
    }
}
